// MultiScalePeriodicPatchEmbedding_57939108823225
// MI455X (gfx1250) — matrix operand order repaired and checked by reading, not compiled, not run on hardware
//
#include <hip/hip_runtime.h>
#include <hip/hip_bf16.h>

// ---------------------------------------------------------------------------
// Compile-time problem geometry
// ---------------------------------------------------------------------------
#define NP 43
#define SEQ 512
#define CIN 14
#define DMODEL 512
#define BATCH 64
#define LTOT 1657           // sum of n_patches over all 43 branches
#define FREQN 256
#define HID 1024            // FREQ * HSF

typedef __attribute__((ext_vector_type(16))) __bf16 v16bf;
typedef __attribute__((ext_vector_type(8)))  float  v8f;

struct WvPtrs { const float* w[NP]; };

struct Meta {
  int ps[NP];                 // patch sizes (ascending, == np.unique order)
  int npat[NP];               // ceil(512/p)
  int rowoff[NP + 1];         // prefix of npat  (concat offset in patch dim)
  int ksteps[NP];             // ceil(14p/32)
  long long aoff[NP + 1];     // packed-A offsets (bf16 elements)
  long long boff[NP + 1];     // packed-B offsets (bf16 elements)
};

constexpr Meta mk() {
  Meta m{};
  constexpr int ps[NP] = {
      2,3,4,5,6,7,8,9,10,11,12,13,14,15,16,17,18,19,20,21,22,
      23,24,25,26,28,30,32,34,36,39,42,46,51,56,64,73,85,102,128,170,256,512};
  int ro = 0; long long ao = 0, bo = 0;
  for (int i = 0; i < NP; ++i) {
    int p = ps[i];
    int n = (SEQ + p - 1) / p;
    int ks = (CIN * p + 31) / 32;
    m.ps[i] = p; m.npat[i] = n; m.ksteps[i] = ks;
    m.rowoff[i] = ro; ro += n;
    m.aoff[i] = ao; ao += (long long)(4 * n) * ks * 512;  // mtiles*ksteps*32lanes*16
    m.boff[i] = bo; bo += 32LL * ks * 512;                // 32 colgroups*ksteps*32*16
  }
  m.rowoff[NP] = ro; m.aoff[NP] = ao; m.boff[NP] = bo;
  return m;
}
constexpr Meta HM = mk();
static_assert(HM.rowoff[NP] == LTOT, "patch-row total mismatch");
static_assert(HM.aoff[NP] % 512 == 0 && HM.boff[NP] % 512 == 0, "align");

__constant__ Meta c_m = mk();

__device__ __forceinline__ unsigned short f2bf(float f) {
  unsigned int u = __builtin_bit_cast(unsigned int, f);
  u += 0x7FFFu + ((u >> 16) & 1u);            // round-to-nearest-even
  return (unsigned short)(u >> 16);
}

// ===========================================================================
// Gates path (f32 VALU; ~10% of total FLOPs)
// ===========================================================================
__global__ void dft_kernel(const float* __restrict__ x,
                           float* __restrict__ xr, float* __restrict__ xi) {
  int idx = blockIdx.x * blockDim.x + threadIdx.x;   // (b*14+c)*256 + (k-1)
  if (idx >= BATCH * CIN * FREQN) return;
  int k  = (idx & (FREQN - 1)) + 1;
  int bc = idx >> 8;
  int b  = bc / CIN;
  int c  = bc - b * CIN;
  const float* xp = x + ((size_t)b * SEQ) * CIN + c;
  const float w = -6.2831853071795864769f * (float)k / (float)SEQ;
  float sr = 0.f, si = 0.f;
  for (int t = 0; t < SEQ; ++t) {
    float s, cs;
    __sincosf(w * (float)t, &s, &cs);
    float v = xp[(size_t)t * CIN];
    sr += v * cs;
    si += v * s;
  }
  xr[idx] = sr;
  xi[idx] = si;
}

__global__ void gates_l1(const float* __restrict__ xr, const float* __restrict__ xi,
                         const float* __restrict__ w1, const float* __restrict__ b1,
                         float* __restrict__ o1r, float* __restrict__ o1i) {
  int idx = blockIdx.x * blockDim.x + threadIdx.x;   // row*1024 + o
  if (idx >= BATCH * CIN * HID) return;
  int o   = idx & (HID - 1);
  int row = idx >> 10;
  const float* rp = xr + (size_t)row * FREQN;
  const float* ip = xi + (size_t)row * FREQN;
  const float* w10 = w1;                      // [256,1024]
  const float* w11 = w1 + (size_t)FREQN * HID;
  float ar = b1[o], ai = b1[HID + o];
  for (int k = 0; k < FREQN; ++k) {
    float a = w10[(size_t)k * HID + o];
    float bi = w11[(size_t)k * HID + o];
    float r = rp[k], im = ip[k];
    ar += r * a - im * bi;
    ai += im * a + r * bi;
  }
  o1r[idx] = fmaxf(ar, 0.f);
  o1i[idx] = fmaxf(ai, 0.f);
}

__global__ void gates_l2(const float* __restrict__ o1r, const float* __restrict__ o1i,
                         const float* __restrict__ w2, const float* __restrict__ b2,
                         float* __restrict__ wts) {
  int idx = blockIdx.x * blockDim.x + threadIdx.x;   // b*43 + pp
  if (idx >= BATCH * NP) return;
  int pp = idx % NP;
  int b  = idx / NP;
  const float* w20 = w2;                      // [1024,43]
  const float* w21 = w2 + (size_t)HID * NP;
  float acc = 0.f;
  for (int c = 0; c < CIN; ++c) {
    const float* r  = o1r + ((size_t)(b * CIN + c) << 10);
    const float* im = o1i + ((size_t)(b * CIN + c) << 10);
    float ar = b2[pp], ai = b2[NP + pp];
    for (int k = 0; k < HID; ++k) {
      float a = w20[(size_t)k * NP + pp];
      float bi = w21[(size_t)k * NP + pp];
      ar += r[k] * a - im[k] * bi;
      ai += im[k] * a + r[k] * bi;
    }
    float sr = copysignf(fmaxf(fabsf(ar) - 0.01f, 0.f), ar);
    float si = copysignf(fmaxf(fabsf(ai) - 0.01f, 0.f), ai);
    acc += sqrtf(sr * sr + si * si);
  }
  wts[idx] = acc * (1.0f / (float)CIN);
}

__global__ void topk_gates(const float* __restrict__ wts, float* __restrict__ gates) {
  int b = blockIdx.x * blockDim.x + threadIdx.x;
  if (b >= BATCH) return;
  float w[NP];
  bool used[NP];
  for (int i = 0; i < NP; ++i) { w[i] = wts[b * NP + i]; used[i] = false; }
  int   idxs[5];
  float vals[5];
  for (int t = 0; t < 5; ++t) {
    int best = 0; float bv = -3.4e38f;
    for (int i = 0; i < NP; ++i)
      if (!used[i] && w[i] > bv) { bv = w[i]; best = i; }
    used[best] = true; idxs[t] = best; vals[t] = bv;
  }
  float mx = vals[0];
  float se = 0.f;
  for (int t = 0; t < 5; ++t) { vals[t] = __expf(vals[t] - mx); se += vals[t]; }
  float inv = 1.0f / se;
  for (int i = 0; i < NP; ++i) gates[b * NP + i] = 0.f;
  for (int t = 0; t < 5; ++t) gates[b * NP + idxs[t]] = vals[t] * inv;
}

// ===========================================================================
// One-time packers: emit bf16 fragments in the exact WMMA VGPR layouts, so the
// GEMM inner loop is pure {vector load -> v_wmma}.
// ===========================================================================

// Packed B: per branch, per 16-col group g (32 of them), per K-step s, per
// lane L: 16 bf16 (32B).  B layout: lane = column, lanes 0-15 K {0-7,16-23},
// lanes 16-31 K {8-15,24-31}, packed pairs per VGPR.
__global__ void pack_b(WvPtrs wv, unsigned short* __restrict__ pb, int total) {
  int q = blockIdx.x * blockDim.x + threadIdx.x;     // 16-element chunk id
  if (q >= total) return;
  int br = 0;
  while (br + 1 < NP && q >= (int)(c_m.boff[br + 1] >> 4)) ++br;
  int l  = q - (int)(c_m.boff[br] >> 4);
  int L  = l & 31;
  int gs = l >> 5;
  int ks = c_m.ksteps[br];
  int s  = gs % ks;
  int g  = gs / ks;
  int K  = CIN * c_m.ps[br];
  int d  = g * 16 + (L & 15);
  int kb = s * 32 + ((L & 16) ? 8 : 0);
  const float* wrow = wv.w[br] + (size_t)d * K;
  union { v16bf v; unsigned short u[16]; } o;
  #pragma unroll
  for (int e = 0; e < 16; ++e) {
    int j = kb + ((e < 8) ? e : (e + 8));
    o.u[e] = f2bf((j < K) ? wrow[j] : 0.0f);
  }
  *reinterpret_cast<v16bf*>(pb + (size_t)q * 16) = o.v;
}

// Packed A: per branch, per M-tile mt, per K-step s, per lane L: 16 bf16.
// A layout: lanes 0-15 row=m0+L, K {0-7,16-23}; lanes 16-31 row=m0+L-16,
// K {8-15,24-31}.  Replication pad (clamp t) and K-tail zero done here, once.
__global__ void pack_a(const float* __restrict__ x, unsigned short* __restrict__ pa,
                       int total) {
  int q = blockIdx.x * blockDim.x + threadIdx.x;
  if (q >= total) return;
  int br = 0;
  while (br + 1 < NP && q >= (int)(c_m.aoff[br + 1] >> 4)) ++br;
  int l  = q - (int)(c_m.aoff[br] >> 4);
  int L  = l & 31;
  int ms = l >> 5;
  int ks = c_m.ksteps[br];
  int s  = ms % ks;
  int mt = ms / ks;
  int p  = c_m.ps[br];
  int ni = c_m.npat[br];
  int K  = CIN * p;
  int arow = mt * 16 + (L & 15);
  int ab = arow / ni;
  int an = arow - ab * ni;
  const float* xb = x + (size_t)ab * SEQ * CIN;
  int kA = (L & 16) ? 8 : 0;
  union { v16bf v; unsigned short u[16]; } o;
  #pragma unroll
  for (int e = 0; e < 16; ++e) {
    int j = s * 32 + ((e < 8) ? (kA + e) : (16 + kA + e - 8));
    int tl = j / CIN;
    int c  = j - tl * CIN;
    int t  = an * p + tl;
    int tt = t > (SEQ - 1) ? (SEQ - 1) : t;
    o.u[e] = f2bf((j < K) ? xb[(size_t)tt * CIN + c] : 0.0f);
  }
  *reinterpret_cast<v16bf*>(pa + (size_t)q * 16) = o.v;
}

// ===========================================================================
// Main GEMM: each wave computes a 16(M) x 64(N) output tile; K-loop body is
// 5 fragment loads (32B each) + 4 v_wmma_f32_16x16x32_bf16.
// Tiles per branch = (64*ni/16) mtiles * 8 ntiles = 32*ni -> prefix 32*rowoff.
// ===========================================================================
__global__ __launch_bounds__(128) void patch_embed_wmma(
    const unsigned short* __restrict__ pa, const unsigned short* __restrict__ pb,
    float* __restrict__ out) {
  const int wave = threadIdx.x >> 5;
  const int lane = threadIdx.x & 31;
  int tile = blockIdx.x * 4 + wave;
  if (tile >= 32 * LTOT) return;              // wave-uniform (exact anyway)

  int br = 0;
  while (br + 1 < NP && tile >= 32 * c_m.rowoff[br + 1]) ++br;
  const int local = tile - 32 * c_m.rowoff[br];
  const int mt = local >> 3;                  // M-tile (16 rows)
  const int nt = local & 7;                   // 64-col group
  const int ks = c_m.ksteps[br];
  const int ni = c_m.npat[br];
  const int m0 = mt * 16;

  const unsigned short* ap = pa + c_m.aoff[br] + ((size_t)mt * ks * 32 + lane) * 16;
  const unsigned short* bp0 = pb + c_m.boff[br] + ((size_t)(nt * 4 + 0) * ks * 32 + lane) * 16;
  const unsigned short* bp1 = pb + c_m.boff[br] + ((size_t)(nt * 4 + 1) * ks * 32 + lane) * 16;
  const unsigned short* bp2 = pb + c_m.boff[br] + ((size_t)(nt * 4 + 2) * ks * 32 + lane) * 16;
  const unsigned short* bp3 = pb + c_m.boff[br] + ((size_t)(nt * 4 + 3) * ks * 32 + lane) * 16;

  v8f a0 = {}, a1 = {}, a2 = {}, a3 = {};
  for (int s = 0; s < ks; ++s) {
    v16bf af = *reinterpret_cast<const v16bf*>(ap);  ap  += 512;
    v16bf b0 = *reinterpret_cast<const v16bf*>(bp0); bp0 += 512;
    v16bf b1 = *reinterpret_cast<const v16bf*>(bp1); bp1 += 512;
    v16bf b2 = *reinterpret_cast<const v16bf*>(bp2); bp2 += 512;
    v16bf b3 = *reinterpret_cast<const v16bf*>(bp3); bp3 += 512;
    a0 = __builtin_amdgcn_wmma_f32_16x16x32_bf16(false, af, false, b0, (short)0, a0, false, false);
    a1 = __builtin_amdgcn_wmma_f32_16x16x32_bf16(false, af, false, b1, (short)0, a1, false, false);
    a2 = __builtin_amdgcn_wmma_f32_16x16x32_bf16(false, af, false, b2, (short)0, a2, false, false);
    a3 = __builtin_amdgcn_wmma_f32_16x16x32_bf16(false, af, false, b3, (short)0, a3, false, false);
  }

  // Epilogue: add sinusoidal positional embedding, store.
  // pe[n, d] = (d even ? sin : cos)(n * exp(-(ln 1e4)/512 * (d & ~1)))
  const int rowoff = c_m.rowoff[br];
  const int rbase = (lane < 16) ? 0 : 8;      // C/D layout: lanes>=16 hold M=8..15
  v8f accs[4] = {a0, a1, a2, a3};
  #pragma unroll
  for (int qn = 0; qn < 4; ++qn) {
    const int d = nt * 64 + qn * 16 + (lane & 15);
    const float div = __expf(-0.017988945859604367f * (float)(d & ~1));
    #pragma unroll
    for (int r = 0; r < 8; ++r) {
      int m = m0 + rbase + r;
      int b = m / ni;
      int n = m - b * ni;
      float s, cc;
      __sincosf(div * (float)n, &s, &cc);
      float pe = (d & 1) ? cc : s;
      out[((size_t)b * LTOT + rowoff + n) * DMODEL + d] = accs[qn][r] + pe;
    }
  }
}

// ---------------------------------------------------------------------------
// Launch
// ---------------------------------------------------------------------------
extern "C" void kernel_launch(void* const* d_in, const int* in_sizes, int n_in,
                              void* d_out, int out_size, void* d_ws, size_t ws_size,
                              hipStream_t stream) {
  (void)in_sizes; (void)n_in; (void)out_size; (void)ws_size;
  const float* x  = (const float*)d_in[0];
  const float* w1 = (const float*)d_in[1];
  const float* b1 = (const float*)d_in[2];
  const float* w2 = (const float*)d_in[3];
  const float* b2 = (const float*)d_in[4];
  WvPtrs wv;
  for (int i = 0; i < NP; ++i) wv.w[i] = (const float*)d_in[5 + i];

  float* out   = (float*)d_out;
  float* gates = out + (size_t)BATCH * LTOT * DMODEL;

  // workspace: f32 gates scratch, then bf16 packed A, then bf16 packed B
  float* ws  = (float*)d_ws;
  float* xr  = ws;
  float* xi  = xr + (size_t)BATCH * CIN * FREQN;
  float* o1r = xi + (size_t)BATCH * CIN * FREQN;
  float* o1i = o1r + (size_t)BATCH * CIN * HID;
  float* wts = o1i + (size_t)BATCH * CIN * HID;
  unsigned short* packA = (unsigned short*)(wts + BATCH * NP);
  unsigned short* packB = packA + HM.aoff[NP];

  constexpr int ACH = (int)(HM.aoff[NP] / 16);   // packed-A 16-elem chunks
  constexpr int BCH = (int)(HM.boff[NP] / 16);   // packed-B 16-elem chunks

  // one-time packers (bf16, exact WMMA fragment layouts)
  pack_a<<<(ACH + 255) / 256, 256, 0, stream>>>(x, packA, ACH);
  pack_b<<<(BCH + 255) / 256, 256, 0, stream>>>(wv, packB, BCH);

  // gates pipeline
  dft_kernel<<<(BATCH * CIN * FREQN + 255) / 256, 256, 0, stream>>>(x, xr, xi);
  gates_l1<<<(BATCH * CIN * HID + 255) / 256, 256, 0, stream>>>(xr, xi, w1, b1, o1r, o1i);
  gates_l2<<<(BATCH * NP + 255) / 256, 256, 0, stream>>>(o1r, o1i, w2, b2, wts);
  topk_gates<<<1, 64, 0, stream>>>(wts, gates);

  // main WMMA GEMM: 32*1657 = 53024 wave-tiles, 4 waves/block -> 13256 blocks
  patch_embed_wmma<<<(32 * LTOT) / 4, 128, 0, stream>>>(packA, packB, out);
}
